// NextGram_27109833572364
// MI455X (gfx1250) — matrix operand order repaired and checked by reading, not compiled, not run on hardware
//
#include <hip/hip_runtime.h>
#include <hip/hip_bf16.h>
#include <stdint.h>

// ---------------------------------------------------------------------------
// Problem dims (fixed by reference)
// ---------------------------------------------------------------------------
#define BATCH 32
#define TSEQ  512
#define EMB   1024
#define VOC   8192
#define H4    4096            // 4 * EMB (i,j,f,o gates)
#define BT    (BATCH * TSEQ)  // 16384 rows

typedef __attribute__((ext_vector_type(16))) __bf16 v16bf;
typedef __attribute__((ext_vector_type(8)))  __bf16 v8bf;
typedef __attribute__((ext_vector_type(8)))  float  v8f;

__device__ __forceinline__ unsigned short f32_to_bf16(float f) {
  union { float f; unsigned u; } x; x.f = f;
  unsigned r = x.u + 0x7FFFu + ((x.u >> 16) & 1u);   // round-to-nearest-even
  return (unsigned short)(r >> 16);
}

__device__ __forceinline__ float sigmoidf_(float x) {
  return 1.0f / (1.0f + __expf(-x));
}

// A-fragment (16-bit A 16x32, ISA 7.12.2): lane 0-15 row M=lane, K {0..7,16..23};
// lane 16-31 row M=lane-16, K {8..15,24..31}. Row-major A => two 16B loads/lane.
__device__ __forceinline__ v16bf load_afrag(const unsigned short* __restrict__ row,
                                            int kbase /* = k0 + (lane>>4)*8 */) {
  v8bf lo = *(const v8bf*)(row + kbase);
  v8bf hi = *(const v8bf*)(row + kbase + 16);
  return __builtin_shufflevector(lo, hi, 0,1,2,3,4,5,6,7,8,9,10,11,12,13,14,15);
}

// ---------------------------------------------------------------------------
// Weight repack: fp32 row-major W[K x N] (row stride ldw) -> bf16 B-fragment
// stream.  Packed element ((nt*KB + kb)*32 + lane)*16 + e  holds
// W[kb*32 + (e>>3)*16 + (lane>>4)*8 + (e&7)][nt*16 + (lane&15)]  (ISA B 32x16 16-bit layout),
// so the GEMM inner loop issues one aligned 32-byte load per lane per k-block.
// ---------------------------------------------------------------------------
__global__ void pack_b_kernel(const float* __restrict__ W, unsigned short* __restrict__ P,
                              int ldw, int K, int N) {
  long long tid = (long long)blockIdx.x * 256 + threadIdx.x;
  long long total = (long long)K * N;
  if (tid >= total) return;
  int e    = (int)(tid & 15);
  int lane = (int)((tid >> 4) & 31);
  long long blk = tid >> 9;           // nt*KB + kb
  int KB = K >> 5;
  int kb = (int)(blk % KB);
  int nt = (int)(blk / KB);
  int k = kb * 32 + (e >> 3) * 16 + (lane >> 4) * 8 + (e & 7);
  int n = nt * 16 + (lane & 15);
  P[tid] = f32_to_bf16(W[(long long)k * ldw + n]);
}

// ---------------------------------------------------------------------------
// Embedding gather + fp32->bf16
// ---------------------------------------------------------------------------
__global__ void embed_kernel(const int* __restrict__ ids, const float* __restrict__ tab,
                             unsigned short* __restrict__ X) {
  long long tid = (long long)blockIdx.x * 256 + threadIdx.x;   // BT*EMB threads
  int col = (int)(tid & (EMB - 1));
  int row = (int)(tid >> 10);
  int tok = ids[row];
  X[tid] = f32_to_bf16(tab[(long long)tok * EMB + col]);
}

// ---------------------------------------------------------------------------
// bf16 GEMM: C[M,N] (fp32, optional bias) = A[M,K] @ Bpacked[K,N]
// 256 threads = 8 waves arranged 2(M) x 4(N); wave tile 64x64 = 4x4 WMMA
// fragments (16 accumulators) -> 16 WMMAs per 16 b128 loads per k-block
// (~32 FLOP/byte of cache traffic, 2x the 32x32 tile).  WG tile 128x256.
// ---------------------------------------------------------------------------
__global__ __launch_bounds__(256, 1) void gemm_bf16_kernel(
    const unsigned short* __restrict__ A, const unsigned short* __restrict__ Bp,
    const float* __restrict__ bias, float* __restrict__ C,
    int M, int N, int K) {
  const int lane = threadIdx.x & 31;
  const int wave = threadIdx.x >> 5;
  const int wm = wave >> 2;            // 0..1
  const int wn = wave & 3;             // 0..3
  const int m0 = blockIdx.y * 128 + wm * 64;
  const int n0 = blockIdx.x * 256 + wn * 64;
  const int hh = lane >> 4;
  const int KB = K >> 5;

  v8f acc[4][4] = {};

  const unsigned short* arow[4];
#pragma unroll
  for (int i = 0; i < 4; ++i)
    arow[i] = A + (long long)(m0 + i * 16 + (lane & 15)) * K;

  const unsigned short* bbase[4];
#pragma unroll
  for (int j = 0; j < 4; ++j)
    bbase[j] = Bp + (((long long)(n0 / 16 + j) * KB) * 32 + lane) * 16;

  for (int kb = 0; kb < KB; ++kb) {
    const int kbase = kb * 32 + hh * 8;
    v16bf af[4], bf[4];
#pragma unroll
    for (int j = 0; j < 4; ++j) {
      __builtin_prefetch(bbase[j] + (long long)(kb + 1) * 512, 0, 3);  // WGP-scope
      bf[j] = *(const v16bf*)(bbase[j] + (long long)kb * 512);
    }
#pragma unroll
    for (int i = 0; i < 4; ++i)
      af[i] = load_afrag(arow[i], kbase);
#pragma unroll
    for (int i = 0; i < 4; ++i)
#pragma unroll
      for (int j = 0; j < 4; ++j)
        acc[i][j] = __builtin_amdgcn_wmma_f32_16x16x32_bf16(
            false, af[i], false, bf[j], (short)0, acc[i][j], false, false);
  }

  const int cm = hh * 8;
  const int cn = lane & 15;
  float bv[4];
#pragma unroll
  for (int j = 0; j < 4; ++j)
    bv[j] = bias ? bias[n0 + j * 16 + cn] : 0.0f;

#pragma unroll
  for (int i = 0; i < 4; ++i) {
#pragma unroll
    for (int v = 0; v < 8; ++v) {
      long long r = m0 + i * 16 + cm + v;
#pragma unroll
      for (int j = 0; j < 4; ++j)
        C[r * N + n0 + j * 16 + cn] = acc[i][j][v] + bv[j];
    }
  }
}

// ---------------------------------------------------------------------------
// Persistent LSTM recurrence.  Grid = 64 WGs (one per 16-hidden-unit block),
// 256 threads = 8 waves: wave = khalf*4 + gate.  Each step:
//   z[:,gate,blk] = h_{t-1} @ Wh (WMMA, K split over 2 half-waves)
//   LDS reduce -> fused gate math (c kept in LDS for all 512 steps)
//   h_t -> bf16 hbuf (next step A) + hout (layer output)
//   agent-scope release/acquire fenced atomic grid barrier.
// Wh (8MB bf16) and hbuf (64KB) stay L2-resident across the whole loop;
// per-step cost is L2 latency + barrier latency, not HBM (23.3 TB/s idle here).
// ---------------------------------------------------------------------------
__global__ __launch_bounds__(256) void lstm_rec_kernel(
    const float* __restrict__ Zx,            // [BT, 4096] precomputed x@Wx
    const unsigned short* __restrict__ Whp,  // packed bf16 [EMB x 4096]
    const float* __restrict__ bias,          // [4096]
    unsigned short* __restrict__ hbuf,       // [32, 1024] bf16, zeroed by host
    unsigned short* __restrict__ hout,       // [BT, 1024] bf16
    unsigned* __restrict__ barrier_cnt) {    // zeroed by host
  __shared__ float zred[8][32][16];
  __shared__ float cstate[32][16];

  const int lane = threadIdx.x & 31;
  const int wave = threadIdx.x >> 5;
  const int gt = wave & 3;          // gate: 0=i 1=j 2=f 3=o
  const int kh = wave >> 2;         // K half
  const int hb = blockIdx.x;        // hidden block (16 units)
  const int hh = lane >> 4;
  const int KB = EMB >> 5;          // 32 k-blocks of 32
  const int nt = gt * 64 + hb;      // N-tile in [0,256)

  for (int e = threadIdx.x; e < 512; e += 256) ((float*)cstate)[e] = 0.0f;
  __syncthreads();

  const unsigned short* arow0 = hbuf + (long long)(lane & 15) * EMB;
  const unsigned short* arow1 = arow0 + (long long)16 * EMB;
  const unsigned short* bbase = Whp + ((long long)nt * KB * 32 + lane) * 16;
  const int elem0 = threadIdx.x * 2;

  for (int t = 0; t < TSEQ; ++t) {
    v8f acc0 = {}, acc1 = {};
    for (int j = 0; j < KB / 2; ++j) {
      int kb = kh * (KB / 2) + j;
      int kbase = kb * 32 + hh * 8;
      __builtin_prefetch(bbase + (long long)(kb + 1) * 512, 0, 3);
      v16bf a0 = load_afrag(arow0, kbase);
      v16bf a1 = load_afrag(arow1, kbase);
      v16bf bf = *(const v16bf*)(bbase + (long long)kb * 512);
      acc0 = __builtin_amdgcn_wmma_f32_16x16x32_bf16(false, a0, false, bf, (short)0, acc0, false, false);
      acc1 = __builtin_amdgcn_wmma_f32_16x16x32_bf16(false, a1, false, bf, (short)0, acc1, false, false);
    }
    for (int v = 0; v < 8; ++v) {                 // C layout: M = v + (lane>>4)*8
      zred[wave][hh * 8 + v][lane & 15]      = acc0[v];
      zred[wave][16 + hh * 8 + v][lane & 15] = acc1[v];
    }
    __syncthreads();

    // fused gate math: 512 (batch, unit) elements over 256 threads
    for (int e = elem0; e < elem0 + 2; ++e) {
      int b = e >> 4, n = e & 15;
      int col = hb * 16 + n;
      long long zr = ((long long)b * TSEQ + t) * H4;
      float zi = zred[0][b][n] + zred[4][b][n] + Zx[zr + col]           + bias[col];
      float zj = zred[1][b][n] + zred[5][b][n] + Zx[zr + EMB + col]     + bias[EMB + col];
      float zf = zred[2][b][n] + zred[6][b][n] + Zx[zr + 2 * EMB + col] + bias[2 * EMB + col];
      float zo = zred[3][b][n] + zred[7][b][n] + Zx[zr + 3 * EMB + col] + bias[3 * EMB + col];
      float c  = cstate[b][n];
      float cn = c * sigmoidf_(zf + 1.0f) + sigmoidf_(zi) * tanhf(zj);  // forget bias 1.0
      float hv = tanhf(cn) * sigmoidf_(zo);
      cstate[b][n] = cn;
      unsigned short hbits = f32_to_bf16(hv);
      hbuf[(long long)b * EMB + col] = hbits;
      hout[((long long)b * TSEQ + t) * EMB + col] = hbits;
    }
    __syncthreads();

    // grid-wide barrier (monotone counter; separate counter per layer)
    if (threadIdx.x == 0) {
      __builtin_amdgcn_fence(__ATOMIC_RELEASE, "agent");
      atomicAdd(barrier_cnt, 1u);
      unsigned target = (unsigned)(t + 1) * gridDim.x;
      volatile unsigned* vc = (volatile unsigned*)barrier_cnt;
      while (*vc < target) __builtin_amdgcn_s_sleep(2);
    }
    __syncthreads();
    __builtin_amdgcn_fence(__ATOMIC_ACQUIRE, "agent");
  }
}

// ---------------------------------------------------------------------------
// Host launch
// ---------------------------------------------------------------------------
extern "C" void kernel_launch(void* const* d_in, const int* in_sizes, int n_in,
                              void* d_out, int out_size, void* d_ws, size_t ws_size,
                              hipStream_t stream) {
  const int*   ids  = (const int*)d_in[0];
  const float* Etab = (const float*)d_in[1];
  const float* W1   = (const float*)d_in[2];
  const float* b1   = (const float*)d_in[3];
  const float* W2   = (const float*)d_in[4];
  const float* b2   = (const float*)d_in[5];
  const float* Wd   = (const float*)d_in[6];
  const float* bd   = (const float*)d_in[7];
  float* out = (float*)d_out;

  char* ws = (char*)d_ws;
  size_t off = 0;
  auto alloc = [&](size_t bytes) -> void* {
    void* p = ws + off;
    off = (off + bytes + 255) & ~(size_t)255;
    return p;
  };
  unsigned short* Xbf  = (unsigned short*)alloc((size_t)BT * EMB * 2);
  unsigned short* h1bf = (unsigned short*)alloc((size_t)BT * EMB * 2);
  unsigned short* h2bf = (unsigned short*)alloc((size_t)BT * EMB * 2);
  unsigned short* W1xp = (unsigned short*)alloc((size_t)EMB * H4 * 2);
  unsigned short* W1hp = (unsigned short*)alloc((size_t)EMB * H4 * 2);
  unsigned short* W2xp = (unsigned short*)alloc((size_t)EMB * H4 * 2);
  unsigned short* W2hp = (unsigned short*)alloc((size_t)EMB * H4 * 2);
  unsigned short* Wdp  = (unsigned short*)alloc((size_t)EMB * VOC * 2);
  float*          Z    = (float*)alloc((size_t)BT * H4 * 4);   // reused for both layers
  unsigned short* hbuf = (unsigned short*)alloc((size_t)32 * EMB * 2);
  unsigned*       cnts = (unsigned*)alloc(256);

  hipMemsetAsync(hbuf, 0, (size_t)32 * EMB * 2, stream);
  hipMemsetAsync(cnts, 0, 256, stream);

  // repack weights -> bf16 B-fragment layout (amortized over whole model)
  int pb = (int)(((long long)EMB * H4) / 256);
  pack_b_kernel<<<pb, 256, 0, stream>>>(W1,                        W1xp, H4, EMB, H4);
  pack_b_kernel<<<pb, 256, 0, stream>>>(W1 + (long long)EMB * H4,  W1hp, H4, EMB, H4);
  pack_b_kernel<<<pb, 256, 0, stream>>>(W2,                        W2xp, H4, EMB, H4);
  pack_b_kernel<<<pb, 256, 0, stream>>>(W2 + (long long)EMB * H4,  W2hp, H4, EMB, H4);
  pack_b_kernel<<<(int)(((long long)EMB * VOC) / 256), 256, 0, stream>>>(Wd, Wdp, VOC, EMB, VOC);

  embed_kernel<<<(int)(((long long)BT * EMB) / 256), 256, 0, stream>>>(ids, Etab, Xbf);

  dim3 gProj(H4 / 256, BT / 128);   // (16, 128)
  dim3 gOut(VOC / 256, BT / 128);   // (32, 128)

  // layer 1: input projection (batched GEMM) then recurrence
  gemm_bf16_kernel<<<gProj, 256, 0, stream>>>(Xbf, W1xp, nullptr, Z, BT, H4, EMB);
  lstm_rec_kernel<<<64, 256, 0, stream>>>(Z, W1hp, b1, hbuf, h1bf, cnts + 0);

  // layer 2
  hipMemsetAsync(hbuf, 0, (size_t)32 * EMB * 2, stream);
  gemm_bf16_kernel<<<gProj, 256, 0, stream>>>(h1bf, W2xp, nullptr, Z, BT, H4, EMB);
  lstm_rec_kernel<<<64, 256, 0, stream>>>(Z, W2hp, b2, hbuf, h2bf, cnts + 16);

  // dense logits
  gemm_bf16_kernel<<<gOut, 256, 0, stream>>>(h2bf, Wdp, bd, out, BT, VOC, EMB);
}
